// LogicalEncoder_58832462021326
// MI455X (gfx1250) — matrix operand order repaired and checked by reading, not compiled, not run on hardware
//
#include <hip/hip_runtime.h>

#define NNODES 20000
#define TSTEPS 64
#define NEDGES 320000
#define E2EDGE (NEDGES + NNODES)
#define DHID   64
#define HEADS  4

typedef __attribute__((ext_vector_type(16))) _Float16 v16h;
typedef __attribute__((ext_vector_type(8)))  float    v8f;

// ---- WMMA fragment index helpers (CDNA5 ISA 7.12.2, 16-bit 16x32 A / 32x16 B) ----
__device__ __forceinline__ int kidxA(int i, int half) {
  // lanes 0-15: K = {0..7,16..23}; lanes 16-31: K = {8..15,24..31}
  return ((i >> 3) << 4) + (half << 3) + (i & 7);
}
__device__ __forceinline__ int kidxB(int i, int half) {
  // lanes 0-15: K = {0..7,16..23}; lanes 16-31: K = {8..15,24..31}
  return ((i >> 3) << 4) + (half << 3) + (i & 7);
}

__device__ __forceinline__ void atomicMaxF(float* addr, float val) {
  unsigned int* ua = (unsigned int*)addr;
  unsigned int old = *ua;
  while (__uint_as_float(old) < val) {
    unsigned int assumed = old;
    old = atomicCAS(ua, assumed, __float_as_uint(val));
    if (old == assumed) break;
  }
}

// -------------------- simple utility kernels --------------------
__global__ void fill_kernel(float* __restrict__ p, float v, long n) {
  long i = (long)blockIdx.x * blockDim.x + threadIdx.x;
  if (i < n) p[i] = v;
}

__global__ void copy_kernel(const float* __restrict__ a, float* __restrict__ b, int n) {
  int i = blockIdx.x * blockDim.x + threadIdx.x;
  if (i < n) b[i] = a[i];
}

__global__ void deg_kernel(const int* __restrict__ ei, float* __restrict__ indeg,
                           float* __restrict__ outdeg, int E) {
  int e = blockIdx.x * blockDim.x + threadIdx.x;
  if (e >= E) return;
  atomicAdd(&outdeg[ei[e]], 1.0f);
  atomicAdd(&indeg[ei[E + e]], 1.0f);
}

__global__ void degmax_kernel(const float* __restrict__ indeg, const float* __restrict__ outdeg,
                              float* __restrict__ red, int N) {
  __shared__ float s0[1024];
  __shared__ float s1[1024];
  int tid = threadIdx.x;
  float m0 = 0.0f, m1 = 0.0f;
  for (int i = tid; i < N; i += 1024) {
    m0 = fmaxf(m0, indeg[i]);
    m1 = fmaxf(m1, outdeg[i]);
  }
  s0[tid] = m0; s1[tid] = m1;
  __syncthreads();
  for (int st = 512; st > 0; st >>= 1) {
    if (tid < st) { s0[tid] = fmaxf(s0[tid], s0[tid + st]); s1[tid] = fmaxf(s1[tid], s1[tid + st]); }
    __syncthreads();
  }
  if (tid == 0) { red[0] = s0[0]; red[1] = s1[0]; }
}

__global__ void topo_kernel(const int* __restrict__ ei, const float* __restrict__ lvlA,
                            float* __restrict__ lvlB, int E) {
  int e = blockIdx.x * blockDim.x + threadIdx.x;
  if (e >= E) return;
  int s = ei[e];
  int d = ei[E + e];
  atomicMaxF(&lvlB[d], lvlA[s] + 1.0f);
}

// comb[n][d] = h_static + h_struct (biases included)
__global__ void comb_kernel(const float* __restrict__ flops, const float* __restrict__ mem,
                            const float* __restrict__ Wst, const float* __restrict__ bst,
                            const float* __restrict__ indeg, const float* __restrict__ outdeg,
                            const float* __restrict__ lvl, const float* __restrict__ red,
                            const float* __restrict__ Wstr, const float* __restrict__ bstr,
                            float* __restrict__ comb, int N) {
  int idx = blockIdx.x * blockDim.x + threadIdx.x;
  if (idx >= N * DHID) return;
  int n = idx >> 6, d = idx & 63;
  float f0 = log1pf(fmaxf(flops[n], 0.0f));
  float f1 = log1pf(fmaxf(mem[n], 0.0f));
  float hs = f0 * Wst[d] + f1 * Wst[DHID + d] + bst[d];
  float s0 = indeg[n] / (red[0] + 1e-6f);
  float s1 = outdeg[n] / (red[1] + 1e-6f);
  float hstr = s0 * Wstr[d] + s1 * Wstr[DHID + d] + lvl[n] * Wstr[2 * DHID + d] + bstr[d];
  comb[idx] = hs + hstr;
}

// -------------------- fused GRU (WMMA recurrent matmul) --------------------
// block: 128 threads (4 waves), 16 nodes; wave w owns output cols [w*48, w*48+48)
__global__ __launch_bounds__(128)
void gru_fuse_kernel(const float* __restrict__ tcs, const float* __restrict__ hls,
                     const float* __restrict__ Wih, const float* __restrict__ Whh,
                     const float* __restrict__ bih, const float* __restrict__ bhh,
                     const float* __restrict__ lng, const float* __restrict__ lnb,
                     const float* __restrict__ comb, float* __restrict__ x0, int N) {
  __shared__ float h_lds[16][68];    // padded to avoid bank conflicts
  __shared__ float gh_lds[16][196];

  const int tid  = threadIdx.x;
  const int wave = tid >> 5;
  const int lane = tid & 31;
  const int r    = lane & 15;
  const int half = lane >> 4;
  const int nodeBase = blockIdx.x * 16;

  // Preload B fragments of Whh^T (constant over all 64 steps): 3 col tiles x 2 K tiles
  v16h bfrag[3][2];
  for (int tt = 0; tt < 3; ++tt) {
    int col = wave * 48 + tt * 16 + r;         // j index in [0,192)
    for (int kk = 0; kk < 2; ++kk) {
      v16h bf;
      for (int i = 0; i < 16; ++i) {
        int k = kk * 32 + kidxB(i, half);      // B[k][col] = Whh[col][k]
        bf[i] = (_Float16)Whh[col * DHID + k];
      }
      bfrag[tt][kk] = bf;
    }
  }

  for (int i = tid; i < 16 * 68; i += 128) ((float*)h_lds)[i] = 0.0f;
  __syncthreads();

  const int ln_node = tid >> 3;  // 0..15 (aligned groups of 8 lanes within a wave)
  const int q       = tid & 7;   // covers dims [q*8, q*8+8)
  const int gnode   = nodeBase + ln_node;

  float hreg[8];
  for (int i = 0; i < 8; ++i) hreg[i] = 0.0f;

  for (int t = 0; t < TSTEPS; ++t) {
    // ---- gh = h @ Whh^T via WMMA ----
    v16h afrag[2];
    for (int kk = 0; kk < 2; ++kk) {
      v16h af;
      for (int i = 0; i < 16; ++i) {
        int k = kk * 32 + kidxA(i, half);
        af[i] = (_Float16)h_lds[r][k];
      }
      afrag[kk] = af;
    }
    for (int tt = 0; tt < 3; ++tt) {
      v8f acc = {};
      acc = __builtin_amdgcn_wmma_f32_16x16x32_f16(false, afrag[0], false, bfrag[tt][0],
                                                   (short)0, acc, false, false);
      acc = __builtin_amdgcn_wmma_f32_16x16x32_f16(false, afrag[1], false, bfrag[tt][1],
                                                   (short)0, acc, false, false);
      int col = wave * 48 + tt * 16 + r;
      for (int vv = 0; vv < 8; ++vv)
        gh_lds[half * 8 + vv][col] = acc[vv];
    }
    __syncthreads();  // gh complete; all h reads done

    // ---- gates (PyTorch r,z,n order) ----
    float xc = tcs[gnode * TSTEPS + t];
    float xl = log1pf(fmaxf(hls[gnode * TSTEPS + t], 0.0f));
    for (int i = 0; i < 8; ++i) {
      int d = q * 8 + i;
      float gir = xc * Wih[d * 2]              + xl * Wih[d * 2 + 1]              + bih[d];
      float giz = xc * Wih[(DHID + d) * 2]     + xl * Wih[(DHID + d) * 2 + 1]     + bih[DHID + d];
      float gin = xc * Wih[(2 * DHID + d) * 2] + xl * Wih[(2 * DHID + d) * 2 + 1] + bih[2 * DHID + d];
      float ghr = gh_lds[ln_node][d]            + bhh[d];
      float ghz = gh_lds[ln_node][DHID + d]     + bhh[DHID + d];
      float ghn = gh_lds[ln_node][2 * DHID + d] + bhh[2 * DHID + d];
      float rg = 1.0f / (1.0f + expf(-(gir + ghr)));
      float zg = 1.0f / (1.0f + expf(-(giz + ghz)));
      float ng = tanhf(gin + rg * ghn);
      hreg[i] = (1.0f - zg) * ng + zg * hreg[i];
    }
    for (int i = 0; i < 8; ++i) h_lds[ln_node][q * 8 + i] = hreg[i];
    __syncthreads();  // h writes visible before next step's A-frag loads
  }

  // ---- LayerNorm over the node's 64 dims (8 aligned lanes per node) ----
  float s = 0.0f, ss = 0.0f;
  for (int i = 0; i < 8; ++i) { s += hreg[i]; ss += hreg[i] * hreg[i]; }
  for (int m = 1; m < 8; m <<= 1) {
    s  += __shfl_xor(s,  m, 32);
    ss += __shfl_xor(ss, m, 32);
  }
  float mean = s / 64.0f;
  float var  = ss / 64.0f - mean * mean;
  float inv  = rsqrtf(var + 1e-5f);
  for (int i = 0; i < 8; ++i) {
    int d = q * 8 + i;
    float hd = (hreg[i] - mean) * inv * lng[d] + lnb[d];
    float v = comb[gnode * DHID + d] + hd;
    x0[gnode * DHID + d] = fmaxf(v, 0.0f);  // relu(h_static + h_dyn + h_struct)
  }
}

// -------------------- generic WMMA GEMM: C[M,N] = A[M,K] @ B[K,N] (f32 in, f16 mma) --------------------
__global__ __launch_bounds__(128)
void gemm16_wmma(const float* __restrict__ A, const float* __restrict__ B,
                 float* __restrict__ C, int M, int N, int K) {
  const int tid  = threadIdx.x;
  const int wave = tid >> 5;
  const int lane = tid & 31;
  const int r    = lane & 15;
  const int half = lane >> 4;
  const int rowBase = blockIdx.x * 16;
  const int colBase = blockIdx.y * 64 + wave * 16;
  v8f acc = {};
  for (int kb = 0; kb < K; kb += 32) {
    v16h a, b;
    for (int i = 0; i < 16; ++i) {
      int ka = kb + kidxA(i, half);
      int kbi = kb + kidxB(i, half);
      a[i] = (_Float16)A[(rowBase + r) * K + ka];
      b[i] = (_Float16)B[kbi * N + colBase + r];
    }
    acc = __builtin_amdgcn_wmma_f32_16x16x32_f16(false, a, false, b, (short)0, acc, false, false);
  }
  for (int vv = 0; vv < 8; ++vv)
    C[(rowBase + half * 8 + vv) * N + colBase + r] = acc[vv];
}

// -------------------- GAT pieces --------------------
__global__ void alpha_kernel(const float* __restrict__ xp, const float* __restrict__ a_s,
                             const float* __restrict__ a_d, float* __restrict__ os,
                             float* __restrict__ od, int N, int H) {
  int idx = blockIdx.x * blockDim.x + threadIdx.x;
  if (idx >= N * H) return;
  int n = idx / H, h = idx - n * H;
  float s = 0.0f, d = 0.0f;
  const float* xv = xp + (n * H + h) * 64;
  for (int c = 0; c < 64; ++c) {
    float v = xv[c];
    s += v * a_s[h * 64 + c];
    d += v * a_d[h * 64 + c];
  }
  os[idx] = s;
  od[idx] = d;
}

__device__ __forceinline__ void edge_sd(const int* ei, int E, int e, int& s, int& d) {
  if (e < E) { s = ei[e]; d = ei[E + e]; }
  else       { s = e - E; d = e - E; }   // GATConv default self-loops
}

__global__ void edge_logit_kernel(const int* __restrict__ ei, int E, int Etot, int H,
                                  const float* __restrict__ as, const float* __restrict__ ad,
                                  float* __restrict__ ebuf, float* __restrict__ m) {
  int idx = blockIdx.x * blockDim.x + threadIdx.x;
  if (idx >= Etot * H) return;
  int e = idx / H, h = idx - e * H;
  int s, d; edge_sd(ei, E, e, s, d);
  float v = as[s * H + h] + ad[d * H + h];
  v = v > 0.0f ? v : 0.2f * v;            // leaky_relu(., 0.2)
  ebuf[idx] = v;
  atomicMaxF(&m[d * H + h], v);
}

__global__ void edge_exp_kernel(const int* __restrict__ ei, int E, int Etot, int H,
                                float* __restrict__ ebuf, const float* __restrict__ m,
                                float* __restrict__ ssum) {
  int idx = blockIdx.x * blockDim.x + threadIdx.x;
  if (idx >= Etot * H) return;
  int e = idx / H, h = idx - e * H;
  int s, d; edge_sd(ei, E, e, s, d);
  (void)s;
  float ex = expf(ebuf[idx] - m[d * H + h]);
  ebuf[idx] = ex;
  atomicAdd(&ssum[d * H + h], ex);
}

__global__ void edge_agg_kernel(const int* __restrict__ ei, int E, int Etot, int H,
                                const float* __restrict__ ebuf, const float* __restrict__ ssum,
                                const float* __restrict__ xp, float* __restrict__ out) {
  int idx = blockIdx.x * blockDim.x + threadIdx.x;
  if (idx >= Etot * H) return;
  int e = idx / H, h = idx - e * H;
  int s, d; edge_sd(ei, E, e, s, d);
  float w = ebuf[idx] / (ssum[d * H + h] + 1e-16f);
  const float* xs = xp + (s * H + h) * 64;
  float* od = out + (d * H + h) * 64;
  for (int c = 0; c < 64; ++c) atomicAdd(&od[c], w * xs[c]);
}

// bias + optional ELU + LayerNorm; one wave per node
__global__ __launch_bounds__(128)
void bias_act_ln_kernel(const float* __restrict__ in, const float* __restrict__ bias,
                        const float* __restrict__ g, const float* __restrict__ b,
                        float* __restrict__ out, int N, int C, int act) {
  int wave = threadIdx.x >> 5;
  int lane = threadIdx.x & 31;
  int node = blockIdx.x * 4 + wave;
  if (node >= N) return;
  int per = C >> 5;  // 8 for C=256, 2 for C=64
  float vals[8];
  float s = 0.0f, ss = 0.0f;
  for (int i = 0; i < per; ++i) {
    int c = lane * per + i;
    float v = in[node * C + c] + bias[c];
    if (act) v = v > 0.0f ? v : expm1f(v);  // elu
    vals[i] = v; s += v; ss += v * v;
  }
  for (int m = 1; m < 32; m <<= 1) {
    s  += __shfl_xor(s,  m, 32);
    ss += __shfl_xor(ss, m, 32);
  }
  float mean = s / (float)C;
  float var  = ss / (float)C - mean * mean;
  float inv  = rsqrtf(var + 1e-5f);
  for (int i = 0; i < per; ++i) {
    int c = lane * per + i;
    out[node * C + c] = (vals[i] - mean) * inv * g[c] + b[c];
  }
}

// -------------------- host driver --------------------
static inline unsigned nb(long n, int tb) { return (unsigned)((n + tb - 1) / tb); }

extern "C" void kernel_launch(void* const* d_in, const int* in_sizes, int n_in,
                              void* d_out, int out_size, void* d_ws, size_t ws_size,
                              hipStream_t stream) {
  (void)in_sizes; (void)n_in; (void)out_size; (void)ws_size;
  const int*   ei       = (const int*)d_in[0];
  const float* flops    = (const float*)d_in[1];
  const float* mem      = (const float*)d_in[2];
  const float* tcs      = (const float*)d_in[3];
  const float* hls      = (const float*)d_in[4];
  const float* W_static = (const float*)d_in[5];
  const float* b_static = (const float*)d_in[6];
  const float* gru_Wih  = (const float*)d_in[7];
  const float* gru_Whh  = (const float*)d_in[8];
  const float* gru_bih  = (const float*)d_in[9];
  const float* gru_bhh  = (const float*)d_in[10];
  const float* lnt_g    = (const float*)d_in[11];
  const float* lnt_b    = (const float*)d_in[12];
  const float* W_struct = (const float*)d_in[13];
  const float* b_struct = (const float*)d_in[14];
  const float* g1_W     = (const float*)d_in[15];
  const float* g1_as    = (const float*)d_in[16];
  const float* g1_ad    = (const float*)d_in[17];
  const float* g1_b     = (const float*)d_in[18];
  const float* g2_W     = (const float*)d_in[19];
  const float* g2_as    = (const float*)d_in[20];
  const float* g2_ad    = (const float*)d_in[21];
  const float* g2_b     = (const float*)d_in[22];
  const float* ln1_g    = (const float*)d_in[23];
  const float* ln1_b    = (const float*)d_in[24];
  const float* ln2_g    = (const float*)d_in[25];
  const float* ln2_b    = (const float*)d_in[26];

  float* w = (float*)d_ws;
  size_t o = 0;
  float* indeg  = w + o; o += NNODES;
  float* outdeg = w + o; o += NNODES;
  float* lvlA   = w + o; o += NNODES;
  float* lvlB   = w + o; o += NNODES;
  float* red    = w + o; o += 16;
  float* comb   = w + o; o += (size_t)NNODES * DHID;
  float* x0     = w + o; o += (size_t)NNODES * DHID;
  float* bigD   = w + o; o += (size_t)NNODES * 256;   // xp1, later h1
  float* bigE   = w + o; o += (size_t)NNODES * 256;   // out1
  float* as1    = w + o; o += (size_t)NNODES * HEADS;
  float* ad1    = w + o; o += (size_t)NNODES * HEADS;
  float* m1     = w + o; o += (size_t)NNODES * HEADS;
  float* s1     = w + o; o += (size_t)NNODES * HEADS;
  float* eb1    = w + o; o += (size_t)E2EDGE * HEADS;
  float* as2    = w + o; o += NNODES;
  float* ad2    = w + o; o += NNODES;
  float* m2     = w + o; o += NNODES;
  float* s2     = w + o; o += NNODES;
  float* eb2    = w + o; o += E2EDGE;

  float* xp1  = bigD;
  float* out1 = bigE;
  float* h1   = bigD;   // xp1 dead after agg1
  float* xp2  = x0;     // x0 dead after gemm1
  float* out2 = comb;   // comb dead after GRU

  const int TB = 256;

  // struct features
  fill_kernel<<<nb(3L * NNODES, TB), TB, 0, stream>>>(indeg, 0.0f, 3L * NNODES);
  deg_kernel<<<nb(NEDGES, TB), TB, 0, stream>>>(ei, indeg, outdeg, NEDGES);
  degmax_kernel<<<1, 1024, 0, stream>>>(indeg, outdeg, red, NNODES);
  float* la = lvlA;
  float* lb = lvlB;
  for (int it = 0; it < 64; ++it) {
    copy_kernel<<<nb(NNODES, TB), TB, 0, stream>>>(la, lb, NNODES);
    topo_kernel<<<nb(NEDGES, TB), TB, 0, stream>>>(ei, la, lb, NEDGES);
    float* t = la; la = lb; lb = t;
  }
  comb_kernel<<<nb((long)NNODES * DHID, TB), TB, 0, stream>>>(
      flops, mem, W_static, b_static, indeg, outdeg, la, red, W_struct, b_struct, comb, NNODES);

  // GRU + fuse -> x0
  gru_fuse_kernel<<<NNODES / 16, 128, 0, stream>>>(
      tcs, hls, gru_Wih, gru_Whh, gru_bih, gru_bhh, lnt_g, lnt_b, comb, x0, NNODES);

  // GAT layer 1
  gemm16_wmma<<<dim3(NNODES / 16, 4), 128, 0, stream>>>(x0, g1_W, xp1, NNODES, 4 * DHID, DHID);
  alpha_kernel<<<nb((long)NNODES * HEADS, TB), TB, 0, stream>>>(xp1, g1_as, g1_ad, as1, ad1, NNODES, HEADS);
  fill_kernel<<<nb((long)NNODES * HEADS, TB), TB, 0, stream>>>(m1, -1e30f, (long)NNODES * HEADS);
  fill_kernel<<<nb((long)NNODES * HEADS, TB), TB, 0, stream>>>(s1, 0.0f, (long)NNODES * HEADS);
  fill_kernel<<<nb((long)NNODES * 256, TB), TB, 0, stream>>>(out1, 0.0f, (long)NNODES * 256);
  edge_logit_kernel<<<nb((long)E2EDGE * HEADS, TB), TB, 0, stream>>>(ei, NEDGES, E2EDGE, HEADS, as1, ad1, eb1, m1);
  edge_exp_kernel<<<nb((long)E2EDGE * HEADS, TB), TB, 0, stream>>>(ei, NEDGES, E2EDGE, HEADS, eb1, m1, s1);
  edge_agg_kernel<<<nb((long)E2EDGE * HEADS, TB), TB, 0, stream>>>(ei, NEDGES, E2EDGE, HEADS, eb1, s1, xp1, out1);
  bias_act_ln_kernel<<<NNODES / 4, 128, 0, stream>>>(out1, g1_b, ln1_g, ln1_b, h1, NNODES, 256, 1);

  // GAT layer 2
  gemm16_wmma<<<dim3(NNODES / 16, 1), 128, 0, stream>>>(h1, g2_W, xp2, NNODES, DHID, 256);
  alpha_kernel<<<nb(NNODES, TB), TB, 0, stream>>>(xp2, g2_as, g2_ad, as2, ad2, NNODES, 1);
  fill_kernel<<<nb(NNODES, TB), TB, 0, stream>>>(m2, -1e30f, NNODES);
  fill_kernel<<<nb(NNODES, TB), TB, 0, stream>>>(s2, 0.0f, NNODES);
  fill_kernel<<<nb((long)NNODES * DHID, TB), TB, 0, stream>>>(out2, 0.0f, (long)NNODES * DHID);
  edge_logit_kernel<<<nb(E2EDGE, TB), TB, 0, stream>>>(ei, NEDGES, E2EDGE, 1, as2, ad2, eb2, m2);
  edge_exp_kernel<<<nb(E2EDGE, TB), TB, 0, stream>>>(ei, NEDGES, E2EDGE, 1, eb2, m2, s2);
  edge_agg_kernel<<<nb(E2EDGE, TB), TB, 0, stream>>>(ei, NEDGES, E2EDGE, 1, eb2, s2, xp2, out2);
  bias_act_ln_kernel<<<NNODES / 4, 128, 0, stream>>>(out2, g2_b, ln2_g, ln2_b, (float*)d_out, NNODES, DHID, 0);
}
